// SwinTransformerBlock3D_17128329576743
// MI455X (gfx1250) — matrix operand order repaired and checked by reading, not compiled, not run on hardware
//
#include <hip/hip_runtime.h>
#include <hip/hip_fp16.h>

typedef __attribute__((ext_vector_type(16))) _Float16 v16h;
typedef __attribute__((ext_vector_type(8)))  float    v8f;
typedef __attribute__((ext_vector_type(4)))  float    f32x4;
typedef __attribute__((ext_vector_type(4)))  unsigned int u32x4;
typedef __attribute__((ext_vector_type(8)))  int      i32x8;
typedef __attribute__((ext_vector_type(4)))  int      i32x4;

// ---------------------------------------------------------------------------
// TDM: 2D f16 tile load Global -> LDS (descriptor per CDNA5 ISA 8.3/8.4).
// 6-arg builtin form (clang-23 / therock-10.0 lane).
// ---------------------------------------------------------------------------
__device__ __forceinline__ void tdm_load_tile_f16(
    unsigned lds_off, const _Float16* g, unsigned tile_w, unsigned tile_h,
    unsigned tensor_w, unsigned tensor_h, unsigned row_stride_elems) {
  unsigned long long ga = (unsigned long long)(uintptr_t)g;
  u32x4 g0;
  g0[0] = 1u;                                   // count=1, user descriptor
  g0[1] = lds_off;                              // lds_addr (bytes)
  g0[2] = (unsigned)ga;                         // global_addr[31:0]
  g0[3] = (unsigned)((ga >> 32) & 0x01FFFFFFu)  // global_addr[56:32]
        | (2u << 30);                           // type = 2 ("image")
  i32x8 g1;
  g1[0] = (int)(1u << 16);                                   // data_size=1 (2B)
  g1[1] = (int)((tensor_w & 0xFFFFu) << 16);                 // tensor_dim0 lo
  g1[2] = (int)((tensor_w >> 16) | ((tensor_h & 0xFFFFu) << 16));
  g1[3] = (int)((tensor_h >> 16) | (tile_w << 16));          // tile_dim0
  g1[4] = (int)(tile_h & 0xFFFFu);                           // tile_dim1 (dim2=0)
  g1[5] = (int)row_stride_elems;                             // dim0_stride lo
  g1[6] = 0;
  g1[7] = 0;
  i32x4 g2 = {0, 0, 0, 0};                                   // group2 (2D: unused)
  i32x4 g3 = {0, 0, 0, 0};                                   // group3 (2D: unused)
  i32x8 g4 = {0, 0, 0, 0, 0, 0, 0, 0};                       // extra group (zero)
  __builtin_amdgcn_tensor_load_to_lds(g0, g1, g2, g3, g4, 0);
}

// ---------------------------------------------------------------------------
// Weight convert: row-major f32 [K][N] -> f16 transposed [N][K]
// ---------------------------------------------------------------------------
__global__ __launch_bounds__(256) void wconv_kernel(const float* __restrict__ w,
                                                    _Float16* __restrict__ wt,
                                                    int K, int N) {
  int idx = blockIdx.x * 256 + threadIdx.x;
  if (idx >= K * N) return;
  int k = idx / N, n = idx - k * N;
  wt[(size_t)n * K + k] = (_Float16)w[idx];
}

// ---------------------------------------------------------------------------
// LayerNorm (one wave per token). mode 0: gather with cyclic shift + window
// partition (Swin roll by -(1,2,2)); mode 1: natural token order (LN2).
// ---------------------------------------------------------------------------
__global__ __launch_bounds__(256) void ln_kernel(const float* __restrict__ src,
                                                 const float* __restrict__ w,
                                                 const float* __restrict__ b,
                                                 _Float16* __restrict__ dst,
                                                 int mode) {
  int tid  = blockIdx.x * 8 + (threadIdx.x >> 5);
  int lane = threadIdx.x & 31;
  size_t srow;
  if (mode == 0) {
    int win = tid >> 5, n = tid & 31;
    int it = n >> 4, ih = (n >> 2) & 3, iw = n & 3;
    int ww_ = win & 15, hw = (win >> 4) & 15, tw = (win >> 8) & 7, bb = win >> 11;
    int t  = (tw * 2 + it + 1) & 15;
    int h  = (hw * 4 + ih + 2) & 63;
    int ww2 = (ww_ * 4 + iw + 2) & 63;
    srow = (((size_t)bb * 16 + t) * 64 + h) * 64 + ww2;
  } else {
    srow = (size_t)tid;
  }
  const float* xr = src + srow * 256;
  float v[8], s = 0.f;
#pragma unroll
  for (int u = 0; u < 8; ++u) { v[u] = xr[lane + u * 32]; s += v[u]; }
#pragma unroll
  for (int off = 1; off < 32; off <<= 1) s += __shfl_xor(s, off);
  float mu = s * (1.f / 256.f);
  float var = 0.f;
#pragma unroll
  for (int u = 0; u < 8; ++u) { float d = v[u] - mu; var += d * d; }
#pragma unroll
  for (int off = 1; off < 32; off <<= 1) var += __shfl_xor(var, off);
  float r = rsqrtf(var * (1.f / 256.f) + 1e-5f);
#pragma unroll
  for (int u = 0; u < 8; ++u) {
    int c = lane + u * 32;
    dst[(size_t)tid * 256 + c] = (_Float16)((v[u] - mu) * r * w[c] + b[c]);
  }
}

// ---------------------------------------------------------------------------
// Generic WMMA f16 GEMM: D[M x N] = A[M x K] * Wt[N x K]^T, epilogue functor.
// Block tile 128x128, 8 waves (16 rows x 128 cols each), K-step 32.
// A tiles staged via TDM into double-buffered LDS.
// ---------------------------------------------------------------------------
template <class Epi>
__global__ __launch_bounds__(256) void gemm_f16_wmma(
    const _Float16* __restrict__ A, const _Float16* __restrict__ Wt,
    int K, Epi epi) {
  __shared__ _Float16 Abuf[2][128 * 32];
  const int wv = threadIdx.x >> 5, lane = threadIdx.x & 31;
  const int hi = lane >> 4, lo = lane & 15;
  const int row0 = blockIdx.y * 128, col0 = blockIdx.x * 128;
  const int ksteps = K >> 5;

  v8f acc[8] = {};

  if (wv == 0)
    tdm_load_tile_f16((unsigned)(uintptr_t)&Abuf[0][0],
                      A + (size_t)row0 * K, 32, 128,
                      (unsigned)K, 128, (unsigned)K);

  for (int ks = 0; ks < ksteps; ++ks) {
    if (wv == 0) __builtin_amdgcn_s_wait_tensorcnt(0);
    __syncthreads();                 // buf(ks&1) ready; prev compute finished
    if (wv == 0 && ks + 1 < ksteps)  // prefetch next K-slab while computing
      tdm_load_tile_f16((unsigned)(uintptr_t)&Abuf[(ks + 1) & 1][0],
                        A + (size_t)row0 * K + (size_t)(ks + 1) * 32, 32, 128,
                        (unsigned)K, 128, (unsigned)K);

    const _Float16* At = &Abuf[ks & 1][0];
    const int lr = wv * 16 + lo;
    v16h a;
    ((f32x4*)&a)[0] = *(const f32x4*)(At + lr * 32 + hi * 16);
    ((f32x4*)&a)[1] = *(const f32x4*)(At + lr * 32 + 8 + hi * 16);
    const _Float16* Bp = Wt + (size_t)(col0 + lo) * K + ks * 32 + hi * 16;
#pragma unroll
    for (int sub = 0; sub < 8; ++sub) {
      v16h bfrag = *(const v16h*)(Bp + (size_t)sub * 16 * K);
      acc[sub] = __builtin_amdgcn_wmma_f32_16x16x32_f16(
          false, a, false, bfrag, (short)0, acc[sub], false, false);
    }
  }

#pragma unroll
  for (int sub = 0; sub < 8; ++sub)
#pragma unroll
    for (int vr = 0; vr < 8; ++vr)
      epi(row0 + wv * 16 + hi * 8 + vr, col0 + sub * 16 + lo, acc[sub][vr]);
}

// ---------------------------------------------------------------------------
// Epilogues
// ---------------------------------------------------------------------------
struct EpiQKV {  // bias + q-scale; scatter q[n][d], k[n][d], v transposed [d][n]
  const float* bias; _Float16 *q, *k, *v;
  __device__ void operator()(int r, int c, float val) const {
    val += bias[c];
    int win = r >> 5, n = r & 31;
    int which = c >> 8, head = (c >> 5) & 7, dd = c & 31;
    size_t base = ((size_t)win * 8 + head) * 1024;
    if (which == 0)      q[base + n * 32 + dd] = (_Float16)(val * 0.17677669529663689f);
    else if (which == 1) k[base + n * 32 + dd] = (_Float16)val;
    else                 v[base + dd * 32 + n] = (_Float16)val;
  }
};

struct EpiProj {  // bias + window-reverse + roll(+1,+2,+2) + residual, f32 out
  const float* bias; const float* x; float* out;
  __device__ void operator()(int r, int c, float val) const {
    int win = r >> 5, n = r & 31;
    int it = n >> 4, ih = (n >> 2) & 3, iw = n & 3;
    int ww_ = win & 15, hw = (win >> 4) & 15, tw = (win >> 8) & 7, bb = win >> 11;
    int t = (tw * 2 + it + 1) & 15;
    int h = (hw * 4 + ih + 2) & 63;
    int w = (ww_ * 4 + iw + 2) & 63;
    size_t off = ((((size_t)bb * 16 + t) * 64 + h) * 64 + w) * 256 + c;
    out[off] = x[off] + val + bias[c];
  }
};

struct EpiMlp1 {  // bias + exact GELU, f16 out
  const float* bias; _Float16* h1;
  __device__ void operator()(int r, int c, float val) const {
    val += bias[c];
    float g = 0.5f * val * (1.0f + erff(val * 0.70710678118654752f));
    h1[(size_t)r * 1024 + c] = (_Float16)g;
  }
};

struct EpiMlp2 {  // bias + residual read-modify-write on d_out
  const float* bias; float* out;
  __device__ void operator()(int r, int c, float val) const {
    size_t off = (size_t)r * 256 + c;
    out[off] = out[off] + val + bias[c];
  }
};

// ---------------------------------------------------------------------------
// Attention: one block per window, one wave per head. N=32, d=32.
// ---------------------------------------------------------------------------
__device__ __forceinline__ float relbias(const float* __restrict__ tbl,
                                         int i, int j, int head) {
  int dt = (i >> 4)       - (j >> 4)       + 1;   // * 49
  int dh = ((i >> 2) & 3) - ((j >> 2) & 3) + 3;   // * 7
  int dw = (i & 3)        - (j & 3)        + 3;
  return tbl[(dt * 49 + dh * 7 + dw) * 8 + head];
}

__global__ __launch_bounds__(256) void attn_kernel(
    const _Float16* __restrict__ qb, const _Float16* __restrict__ kb,
    const _Float16* __restrict__ vb, const float* __restrict__ rel,
    _Float16* __restrict__ ob) {
  __shared__ _Float16 plds[8][32 * 32];
  const int win = blockIdx.x;
  const int head = threadIdx.x >> 5;
  const int lane = threadIdx.x & 31;
  const int hi = lane >> 4, lo = lane & 15;
  const size_t base = ((size_t)win * 8 + head) * 1024;
  const _Float16* qh = qb + base;
  const _Float16* kh = kb + base;
  const _Float16* vh = vb + base;

  // q as A-fragments (rows), k as B-fragments (cols of K^T, contiguous in d)
  v16h qa[2], kf[2];
#pragma unroll
  for (int t = 0; t < 2; ++t) {
    int i = t * 16 + lo;
    ((f32x4*)&qa[t])[0] = *(const f32x4*)(qh + i * 32 + hi * 16);
    ((f32x4*)&qa[t])[1] = *(const f32x4*)(qh + i * 32 + 8 + hi * 16);
    kf[t] = *(const v16h*)(kh + i * 32 + hi * 16);
  }
  v8f S[2][2] = {};
#pragma unroll
  for (int ti = 0; ti < 2; ++ti)
#pragma unroll
    for (int tj = 0; tj < 2; ++tj)
      S[ti][tj] = __builtin_amdgcn_wmma_f32_16x16x32_f16(
          false, qa[ti], false, kf[tj], (short)0, S[ti][tj], false, false);

  // relative-position bias + exact softmax (row lives in one 16-lane half)
#pragma unroll
  for (int ti = 0; ti < 2; ++ti) {
#pragma unroll
    for (int vr = 0; vr < 8; ++vr) {
      int i = ti * 16 + hi * 8 + vr;
      float s0 = S[ti][0][vr] + relbias(rel, i, lo, head);
      float s1 = S[ti][1][vr] + relbias(rel, i, 16 + lo, head);
      float m = fmaxf(s0, s1);
#pragma unroll
      for (int off = 1; off < 16; off <<= 1) m = fmaxf(m, __shfl_xor(m, off));
      float e0 = __expf(s0 - m), e1 = __expf(s1 - m);
      float sm = e0 + e1;
#pragma unroll
      for (int off = 1; off < 16; off <<= 1) sm += __shfl_xor(sm, off);
      float inv = 1.0f / sm;
      plds[head][i * 32 + lo]      = (_Float16)(e0 * inv);
      plds[head][i * 32 + 16 + lo] = (_Float16)(e1 * inv);
    }
  }

  // O = P @ V : P re-loaded as A-fragments from LDS, V (stored [d][n]) as B
  v16h pf[2], vf[2];
  const _Float16* pr = &plds[head][0];
#pragma unroll
  for (int t = 0; t < 2; ++t) {
    int i = t * 16 + lo;
    ((f32x4*)&pf[t])[0] = *(const f32x4*)(pr + i * 32 + hi * 16);
    ((f32x4*)&pf[t])[1] = *(const f32x4*)(pr + i * 32 + 8 + hi * 16);
    vf[t] = *(const v16h*)(vh + i * 32 + hi * 16);
  }
  v8f O[2][2] = {};
#pragma unroll
  for (int ti = 0; ti < 2; ++ti)
#pragma unroll
    for (int tc = 0; tc < 2; ++tc)
      O[ti][tc] = __builtin_amdgcn_wmma_f32_16x16x32_f16(
          false, pf[ti], false, vf[tc], (short)0, O[ti][tc], false, false);

  _Float16* orow = ob + (size_t)win * 32 * 256 + head * 32;
#pragma unroll
  for (int ti = 0; ti < 2; ++ti)
#pragma unroll
    for (int tc = 0; tc < 2; ++tc)
#pragma unroll
      for (int vr = 0; vr < 8; ++vr)
        orow[(size_t)(ti * 16 + hi * 8 + vr) * 256 + tc * 16 + lo] =
            (_Float16)O[ti][tc][vr];
}

// ---------------------------------------------------------------------------
// Host launcher
// ---------------------------------------------------------------------------
extern "C" void kernel_launch(void* const* d_in, const int* in_sizes, int n_in,
                              void* d_out, int out_size, void* d_ws, size_t ws_size,
                              hipStream_t stream) {
  (void)in_sizes; (void)n_in; (void)out_size; (void)ws_size;
  const float* x       = (const float*)d_in[0];
  const float* n1w     = (const float*)d_in[1];
  const float* n1b     = (const float*)d_in[2];
  const float* qkv_w   = (const float*)d_in[3];
  const float* qkv_b   = (const float*)d_in[4];
  const float* proj_w  = (const float*)d_in[5];
  const float* proj_b  = (const float*)d_in[6];
  const float* rel_tab = (const float*)d_in[7];
  const float* n2w     = (const float*)d_in[8];
  const float* n2b     = (const float*)d_in[9];
  const float* w1      = (const float*)d_in[10];
  const float* b1      = (const float*)d_in[11];
  const float* w2      = (const float*)d_in[12];
  const float* b2      = (const float*)d_in[13];
  float* out = (float*)d_out;

  char* p = (char*)d_ws;
  auto take = [&](size_t bytes) {
    char* r = p; p += (bytes + 255) & ~(size_t)255; return (void*)r;
  };
  _Float16* wt_qkv = (_Float16*)take((size_t)256 * 768 * 2);
  _Float16* wt_prj = (_Float16*)take((size_t)256 * 256 * 2);
  _Float16* wt_m1  = (_Float16*)take((size_t)256 * 1024 * 2);
  _Float16* wt_m2  = (_Float16*)take((size_t)1024 * 256 * 2);
  _Float16* Xw  = (_Float16*)take((size_t)131072 * 256 * 2);  // reused as attn-out
  _Float16* qbf = (_Float16*)take((size_t)4096 * 8192 * 2);   // reused as LN2-out
  _Float16* kbf = (_Float16*)take((size_t)4096 * 8192 * 2);
  _Float16* vbf = (_Float16*)take((size_t)4096 * 8192 * 2);
  _Float16* h1  = (_Float16*)take((size_t)131072 * 1024 * 2);

  // weights -> f16 transposed [N][K]
  wconv_kernel<<<(256 * 768  + 255) / 256, 256, 0, stream>>>(qkv_w,  wt_qkv, 256, 768);
  wconv_kernel<<<(256 * 256  + 255) / 256, 256, 0, stream>>>(proj_w, wt_prj, 256, 256);
  wconv_kernel<<<(256 * 1024 + 255) / 256, 256, 0, stream>>>(w1,     wt_m1,  256, 1024);
  wconv_kernel<<<(1024 * 256 + 255) / 256, 256, 0, stream>>>(w2,     wt_m2, 1024, 256);

  // LN1 + cyclic shift + window partition
  ln_kernel<<<16384, 256, 0, stream>>>(x, n1w, n1b, Xw, 0);
  // QKV projection
  gemm_f16_wmma<<<dim3(6, 1024), 256, 0, stream>>>(
      Xw, wt_qkv, 256, EpiQKV{qkv_b, qbf, kbf, vbf});
  // windowed attention (writes attn output into Xw region, f16 [tok][256])
  attn_kernel<<<4096, 256, 0, stream>>>(qbf, kbf, vbf, rel_tab, Xw);
  // output projection + window reverse + roll + residual  -> d_out (x2, f32)
  gemm_f16_wmma<<<dim3(2, 1024), 256, 0, stream>>>(
      Xw, wt_prj, 256, EpiProj{proj_b, x, out});
  // LN2 (natural order) -> f16 (reuse q buffer)
  ln_kernel<<<16384, 256, 0, stream>>>(out, n2w, n2b, qbf, 1);
  // MLP
  gemm_f16_wmma<<<dim3(8, 1024), 256, 0, stream>>>(
      qbf, wt_m1, 256, EpiMlp1{b1, h1});
  gemm_f16_wmma<<<dim3(2, 1024), 256, 0, stream>>>(
      h1, wt_m2, 1024, EpiMlp2{b2, out});
}
